// KNNCrossAttentionBlock_22668837388561
// MI455X (gfx1250) — matrix operand order repaired and checked by reading, not compiled, not run on hardware
//
#include <hip/hip_runtime.h>
#include <hip/hip_bf16.h>

// ---------------- constants ----------------
constexpr int BATCH = 4;
constexpr int NPTS  = 4096;
constexpr int CDIM  = 128;
constexpr int KNNK  = 16;
constexpr int ROWS  = BATCH * NPTS;              // 16384
constexpr float ATT_SCALE = 0.08838834764831845f; // 1/sqrt(128)

#ifndef __has_builtin
#define __has_builtin(x) 0
#endif
#if __has_builtin(__builtin_amdgcn_global_load_async_to_lds_b128) && \
    __has_builtin(__builtin_amdgcn_s_wait_asynccnt)
#define USE_ASYNC_LDS 1
#else
#define USE_ASYNC_LDS 0
#endif

// ---------------- CDNA5 WMMA plumbing ----------------
typedef __attribute__((ext_vector_type(16))) __bf16 v16bf;
typedef __attribute__((ext_vector_type(8)))  float  v8f;
typedef __attribute__((__vector_size__(16))) int    vi4;   // clang-native int4

union Frag { v16bf v; uint4 q[2]; };

__device__ __forceinline__ unsigned short f2bf(float f) {
    unsigned int u = __float_as_uint(f);
    unsigned int r = (u + 0x7fffu + ((u >> 16) & 1u)) >> 16;
    return (unsigned short)r;
}
__device__ __forceinline__ float bf2f(unsigned short h) {
    return __uint_as_float(((unsigned int)h) << 16);
}

// A fragment: 16x32 bf16, rows m0..m0+15, K slice k0..k0+31, row-major source (stride ld)
__device__ __forceinline__ v16bf load_a_frag(const unsigned short* base, int ld, int m0, int k0) {
    int lane = threadIdx.x & 31;
    const unsigned short* p = base + (m0 + (lane & 15)) * ld + k0 + ((lane >> 4) << 3);
    Frag f;
    f.q[0] = *(const uint4*)p;
    f.q[1] = *(const uint4*)(p + 16);
    return f.v;
}
// B fragment: 32x16 bf16 (KxN), source stored [n][k] row-major (stride ld) -> same k order as A fragment
__device__ __forceinline__ v16bf load_b_frag(const unsigned short* base, int ld, int n0, int k0) {
    int lane = threadIdx.x & 31;
    const unsigned short* p = base + (n0 + (lane & 15)) * ld + k0 + ((lane >> 4) << 3);
    Frag f;
    f.q[0] = *(const uint4*)p;
    f.q[1] = *(const uint4*)(p + 16);
    return f.v;
}
__device__ __forceinline__ v8f wmma_bf16(v16bf a, v16bf b, v8f c) {
    return __builtin_amdgcn_wmma_f32_16x16x32_bf16(false, a, false, b, (short)0, c, false, false);
}

// register-resident guarded top-16 insertion (ascending distance)
__device__ __forceinline__ void topk_insert(float (&bd)[16], int (&bi)[16], float d, int j) {
    if (d < bd[15]) {
        float cd = d; int ci = j;
#pragma unroll
        for (int s = 0; s < 16; ++s) {
            bool lt = cd < bd[s];
            float td = bd[s]; int ti = bi[s];
            if (lt) { bd[s] = cd; bi[s] = ci; cd = td; ci = ti; }
        }
    }
}

// async global->LDS copy of `bytes` (multiple of 4096) by 256 threads, 16B chunks
__device__ __forceinline__ void stage_to_lds(const void* gsrc, void* lds, int bytes) {
    const int t = threadIdx.x;
    const int chunks = bytes >> 4;             // 16B per lane per issue
#if USE_ASYNC_LDS
    typedef __attribute__((address_space(1))) vi4 gvi4;
    typedef __attribute__((address_space(3))) vi4 lvi4;
    gvi4* g = (gvi4*)gsrc;
    lvi4* l = (lvi4*)lds;
    for (int i = t; i < chunks; i += 256)
        __builtin_amdgcn_global_load_async_to_lds_b128(g + i, l + i, 0, 0);
    __builtin_amdgcn_s_wait_asynccnt(0);
#else
    const uint4* g = (const uint4*)gsrc;
    uint4* l = (uint4*)lds;
    for (int i = t; i < chunks; i += 256) l[i] = g[i];
#endif
}

// ---------------- prep: M1 = Wq^T Wk (stored [c][k] bf16), bf16 weight casts ----------------
__global__ __launch_bounds__(256) void prep_weights_kernel(
    const float* Wq0, const float* Wk0, const float* Wq1, const float* Wk1,
    const float* Wv0, const float* Wv1, const float* Wl, const float* Wp1,
    unsigned short* M1t0, unsigned short* M1t1, unsigned short* Wv0b,
    unsigned short* Wv1b, unsigned short* Wlb, unsigned short* Wp1b) {
    int id = blockIdx.x * 256 + threadIdx.x;
    int task = id >> 14;
    int e = id & 16383;
    if (task <= 1) {
        const float* Wq = task ? Wq1 : Wq0;
        const float* Wk = task ? Wk1 : Wk0;
        int c = e >> 7, k = e & 127;
        float s = 0.f;
        for (int o = 0; o < 128; ++o) s += Wq[o * 128 + k] * Wk[o * 128 + c];
        (task ? M1t1 : M1t0)[c * 128 + k] = f2bf(s);
    } else {
        const float* s; unsigned short* d;
        switch (task) {
            case 2:  s = Wv0; d = Wv0b; break;
            case 3:  s = Wv1; d = Wv1b; break;
            case 4:  s = Wl;  d = Wlb;  break;
            default: s = Wp1; d = Wp1b; break;
        }
        d[e] = f2bf(s[e]);
    }
}

// ---------------- P1 = src @ Wp0^T  (K=3, pure VALU) ----------------
__global__ __launch_bounds__(256) void p1_kernel(const float* src, const float* Wp0, float* P1) {
    int id = blockIdx.x * 256 + threadIdx.x;   // ROWS*128 threads
    int r = id >> 7, c = id & 127;
    float v = src[r * 3 + 0] * Wp0[c * 3 + 0]
            + src[r * 3 + 1] * Wp0[c * 3 + 1]
            + src[r * 3 + 2] * Wp0[c * 3 + 2];
    P1[id] = v;
}

// ---------------- KNN stage 1: 3-d coordinates, one thread per row ----------------
__global__ __launch_bounds__(256) void knn_coord_kernel(const float* src, int* idx1) {
    int id = blockIdx.x * 256 + threadIdx.x;   // ROWS threads
    int b = id >> 12;                          // NPTS = 4096
    int i = id & (NPTS - 1);
    const float* sb = src + (size_t)b * NPTS * 3;
    float sx = sb[i * 3 + 0], sy = sb[i * 3 + 1], sz = sb[i * 3 + 2];
    float bd[16]; int bi[16];
#pragma unroll
    for (int s = 0; s < 16; ++s) { bd[s] = 3.4e38f; bi[s] = 0; }
    for (int j = 0; j < NPTS; ++j) {
        float dx = sb[j * 3 + 0] - sx;
        float dy = sb[j * 3 + 1] - sy;
        float dz = sb[j * 3 + 2] - sz;
        float d2 = dx * dx + dy * dy + dz * dz;
        topk_insert(bd, bi, d2, j);
    }
#pragma unroll
    for (int s = 0; s < 16; ++s) idx1[id * 16 + s] = bi[s];
}

// ---------------- bf16 cast + squared-norm for feature-space KNN ----------------
__global__ __launch_bounds__(256) void cast_norm_kernel(const float* x, unsigned short* xb, float* r2) {
    int row = blockIdx.x * 256 + threadIdx.x;  // ROWS threads
    const float* xr = x + (size_t)row * 128;
    float s = 0.f;
    for (int c = 0; c < 128; c += 4) {
        float4 v = *(const float4*)(xr + c);
        s += v.x * v.x + v.y * v.y + v.z * v.z + v.w * v.w;
        xb[row * 128 + c + 0] = f2bf(v.x);
        xb[row * 128 + c + 1] = f2bf(v.y);
        xb[row * 128 + c + 2] = f2bf(v.z);
        xb[row * 128 + c + 3] = f2bf(v.w);
    }
    r2[row] = s;
}

// ---------------- KNN stage 2: streaming Gram-matrix via WMMA + running top-16 ----------------
__global__ __launch_bounds__(256) void knn_feat_kernel(const unsigned short* xb, const float* r2, int* idx2) {
    const int wg = blockIdx.x;                 // BATCH * (NPTS/16) = 1024
    const int b = wg >> 8;
    const int rt = wg & 255;
    const int row0 = b * NPTS + rt * 16;
    const int t = threadIdx.x, wave = t >> 5, lane = t & 31;

    __shared__ alignas(16) unsigned short arow[16 * 128];
    __shared__ alignas(16) unsigned short cb[128 * 128];
    __shared__ float r2row[16];
    __shared__ float r2col[128];
    __shared__ float d2s[8][16][16];
    __shared__ float mg_d[8][16][16];
    __shared__ int   mg_i[8][16][16];

    // stage the 16-row A block (bf16, async to LDS) + its norms
    stage_to_lds(xb + (size_t)row0 * 128, arow, 16 * 128 * 2);
    if (t < 16) r2row[t] = r2[row0 + t];

    float bd[16]; int bi[16];
#pragma unroll
    for (int s = 0; s < 16; ++s) { bd[s] = 3.4e38f; bi[s] = 0; }
    __syncthreads();

    for (int blk = 0; blk < NPTS / 128; ++blk) {
        __syncthreads();
        // stage the 128-point column block (bf16, [point][feat] == B's [n][k])
        stage_to_lds(xb + ((size_t)b * NPTS + blk * 128) * 128, cb, 128 * 128 * 2);
        if (t < 128) r2col[t] = r2[b * NPTS + blk * 128 + t];
        if (blk + 1 < NPTS / 128)
            __builtin_prefetch(xb + ((size_t)b * NPTS + (blk + 1) * 128) * 128);
        __syncthreads();

        v8f acc = {};
#pragma unroll
        for (int kt = 0; kt < 4; ++kt) {
            v16bf a = load_a_frag(arow, 128, 0, kt * 32);
            v16bf bb = load_b_frag(cb, 128, wave * 16, kt * 32);
            acc = wmma_bf16(a, bb, acc);
        }
        int nn = lane & 15;
        int mb = (lane < 16) ? 0 : 8;
#pragma unroll
        for (int v = 0; v < 8; ++v) {
            int m = mb + v;
            float d2 = r2row[m] + r2col[wave * 16 + nn] - 2.0f * acc[v];
            d2s[wave][m][nn] = d2;
        }
        __builtin_amdgcn_wave_barrier();       // wave32 lockstep, in-order DS: safe transpose
        if (lane < 16) {
            int m = lane;
#pragma unroll
            for (int n2 = 0; n2 < 16; ++n2) {
                float d = d2s[wave][m][n2];
                int j = blk * 128 + wave * 16 + n2;
                topk_insert(bd, bi, d, j);
            }
        }
    }

    if (lane < 16) {
#pragma unroll
        for (int s = 0; s < 16; ++s) { mg_d[wave][lane][s] = bd[s]; mg_i[wave][lane][s] = bi[s]; }
    }
    __syncthreads();
    if (t < 16) {                              // merge 8 per-wave lists per row
        int m = t;
        float fd[16]; int fi[16];
#pragma unroll
        for (int s = 0; s < 16; ++s) { fd[s] = 3.4e38f; fi[s] = 0; }
        for (int w2 = 0; w2 < 8; ++w2)
#pragma unroll
            for (int s = 0; s < 16; ++s)
                topk_insert(fd, fi, mg_d[w2][m][s], mg_i[w2][m][s]);
#pragma unroll
        for (int s = 0; s < 16; ++s) idx2[(size_t)(row0 + m) * 16 + s] = fi[s];
    }
}

// ---------------- generic  Y = X @ W^T  (R x 128 @ 128 x 128), W stored [n][k] bf16 ----------------
__global__ __launch_bounds__(256) void gemm_rows_kernel(const float* X, const unsigned short* WBN, float* Y) {
    const int row0 = blockIdx.x * 64;
    const int t = threadIdx.x, wave = t >> 5, lane = t & 31;
    __shared__ alignas(16) unsigned short xa[64 * 128];
    {
        int r = t >> 2, c0 = (t & 3) * 32;
        const float* xr = X + (size_t)(row0 + r) * 128;
        for (int i = 0; i < 32; ++i) xa[r * 128 + c0 + i] = f2bf(xr[c0 + i]);
    }
    __syncthreads();
    int mtile = wave >> 1;
    int nbase = (wave & 1) * 4;
    int nn = lane & 15;
    int mb = (lane < 16) ? 0 : 8;
#pragma unroll
    for (int nt = 0; nt < 4; ++nt) {
        v8f acc = {};
#pragma unroll
        for (int kt = 0; kt < 4; ++kt) {
            v16bf a = load_a_frag(xa, 128, mtile * 16, kt * 32);
            v16bf bb = load_b_frag(WBN, 128, (nbase + nt) * 16, kt * 32);
            acc = wmma_bf16(a, bb, acc);
        }
#pragma unroll
        for (int v = 0; v < 8; ++v)
            Y[(size_t)(row0 + mtile * 16 + mb + v) * 128 + (nbase + nt) * 16 + nn] = acc[v];
    }
}

// ---------------- fused KNN cross-attention (both stages) ----------------
// scores = kv . (X @ M1),  out = (softmax . kv) @ Wv^T,  kv = P_q - P_nb + bias + feat_nb
__global__ __launch_bounds__(256) void attn_kernel(
    const float* feat, const float* P, const int* idx,
    const unsigned short* M1t, const unsigned short* WvBN,
    const float* bias, float* outp) {
    const int row0 = blockIdx.x * 16;          // ROWS/16 = 1024 blocks
    const int b = row0 / NPTS;
    const int t = threadIdx.x, wave = t >> 5, lane = t & 31;

    __shared__ alignas(16) unsigned short xq[16 * 128];
    __shared__ alignas(16) float qk[16 * 128];
    __shared__ alignas(16) unsigned short kvh[128 * 128];   // half tile (8 queries x 16 nbrs)
    __shared__ float sc[128];
    __shared__ float aw[128];
    __shared__ float bsh[128];
    __shared__ int   nidx[256];
    __shared__ alignas(16) unsigned short ctx[16 * 128];

    nidx[t] = idx[(size_t)row0 * 16 + t];
    if (t < 128) bsh[t] = bias[t];
    {
        int q = t >> 4, c0 = (t & 15) * 8;
        const float* fr = feat + (size_t)(row0 + q) * 128;
#pragma unroll
        for (int i = 0; i < 8; ++i) xq[q * 128 + c0 + i] = f2bf(fr[c0 + i]);
    }
    __syncthreads();

    // Qk = Xq @ M1 (each wave owns one 16-col tile)
    {
        v8f acc = {};
#pragma unroll
        for (int kt = 0; kt < 4; ++kt) {
            v16bf a = load_a_frag(xq, 128, 0, kt * 32);
            v16bf bb = load_b_frag(M1t, 128, wave * 16, kt * 32);
            acc = wmma_bf16(a, bb, acc);
        }
        int nn = lane & 15;
        int mb = (lane < 16) ? 0 : 8;
#pragma unroll
        for (int v = 0; v < 8; ++v) qk[(mb + v) * 128 + wave * 16 + nn] = acc[v];
    }
    __syncthreads();

    for (int h = 0; h < 2; ++h) {              // two halves of 8 queries (LDS budget)
        int qb = h * 8;
        {   // build kv rows (gather + precomputed residual projection)
            int r = t >> 1, cw = (t & 1) * 64;
            int q = qb + (r >> 4), j = r & 15;
            int nb = nidx[q * 16 + j];
            const float* Pq = P + (size_t)(row0 + q) * 128;
            const float* Pn = P + ((size_t)b * NPTS + nb) * 128;
            const float* Fn = feat + ((size_t)b * NPTS + nb) * 128;
            for (int c = cw; c < cw + 64; c += 4) {
                float4 a4 = *(const float4*)(Pq + c);
                float4 b4 = *(const float4*)(Pn + c);
                float4 f4 = *(const float4*)(Fn + c);
                kvh[r * 128 + c + 0] = f2bf(a4.x - b4.x + bsh[c + 0] + f4.x);
                kvh[r * 128 + c + 1] = f2bf(a4.y - b4.y + bsh[c + 1] + f4.y);
                kvh[r * 128 + c + 2] = f2bf(a4.z - b4.z + bsh[c + 2] + f4.z);
                kvh[r * 128 + c + 3] = f2bf(a4.w - b4.w + bsh[c + 3] + f4.w);
            }
        }
        __syncthreads();
        if (t < 128) {                         // scores = kv . Qk
            int r = t, q = qb + (r >> 4);
            float s = 0.f;
            for (int c = 0; c < 128; ++c) s += bf2f(kvh[r * 128 + c]) * qk[q * 128 + c];
            sc[r] = s * ATT_SCALE;
        }
        __syncthreads();
        if (t < 8) {                           // softmax over 16 neighbors
            int base = t * 16;
            float mx = sc[base];
#pragma unroll
            for (int j = 1; j < 16; ++j) mx = fmaxf(mx, sc[base + j]);
            float sum = 0.f;
#pragma unroll
            for (int j = 0; j < 16; ++j) { float e = __expf(sc[base + j] - mx); aw[base + j] = e; sum += e; }
            float inv = 1.0f / sum;
#pragma unroll
            for (int j = 0; j < 16; ++j) aw[base + j] *= inv;
        }
        __syncthreads();
        {   // ctx = a^T kv  (aggregate before V projection)
            int q8 = t >> 5, part = t & 31, c0 = part * 4;
#pragma unroll
            for (int cc = 0; cc < 4; ++cc) {
                float s = 0.f;
#pragma unroll
                for (int j = 0; j < 16; ++j)
                    s += aw[q8 * 16 + j] * bf2f(kvh[(q8 * 16 + j) * 128 + c0 + cc]);
                ctx[(qb + q8) * 128 + c0 + cc] = f2bf(s);
            }
        }
        __syncthreads();
    }

    // out = ctx @ Wv^T
    {
        v8f acc = {};
#pragma unroll
        for (int kt = 0; kt < 4; ++kt) {
            v16bf a = load_a_frag(ctx, 128, 0, kt * 32);
            v16bf bb = load_b_frag(WvBN, 128, wave * 16, kt * 32);
            acc = wmma_bf16(a, bb, acc);
        }
        int nn = lane & 15;
        int mb = (lane < 16) ? 0 : 8;
#pragma unroll
        for (int v = 0; v < 8; ++v)
            outp[(size_t)(row0 + mb + v) * 128 + wave * 16 + nn] = acc[v];
    }
}

// ---------------- residual + LN + Linear + LN + transpose ----------------
__global__ __launch_bounds__(256) void post_kernel(
    const float* tgt, const float* o2, const float* g0, const float* b0,
    const unsigned short* WlBN, const float* bl,
    const float* g1, const float* b1, float* outg) {
    const int row0 = blockIdx.x * 16;
    const int b = row0 / NPTS;
    const int n0 = row0 % NPTS;
    const int t = threadIdx.x, wave = t >> 5, lane = t & 31;

    __shared__ float xb_[16 * 128];
    __shared__ alignas(16) unsigned short ha[16 * 128];
    __shared__ float red1[16][16], red2[16][16];
    __shared__ float mu_[16], rv_[16];

    int q = t >> 4, part = t & 15, c0 = part * 8;
    {   // x = tgt + attn_out ; partial LN stats
        float s1 = 0.f, s2 = 0.f;
        const float* tr = tgt + (size_t)(row0 + q) * 128;
        const float* orow = o2 + (size_t)(row0 + q) * 128;
#pragma unroll
        for (int i = 0; i < 8; ++i) {
            float x = tr[c0 + i] + orow[c0 + i];
            xb_[q * 128 + c0 + i] = x;
            s1 += x; s2 += x * x;
        }
        red1[q][part] = s1; red2[q][part] = s2;
    }
    __syncthreads();
    if (t < 16) {
        float s1 = 0.f, s2 = 0.f;
#pragma unroll
        for (int i = 0; i < 16; ++i) { s1 += red1[t][i]; s2 += red2[t][i]; }
        float mu = s1 * (1.0f / 128.0f);
        float var = s2 * (1.0f / 128.0f) - mu * mu;
        mu_[t] = mu; rv_[t] = rsqrtf(var + 1e-5f);
    }
    __syncthreads();
#pragma unroll
    for (int i = 0; i < 8; ++i) {
        int c = c0 + i;
        float v = (xb_[q * 128 + c] - mu_[q]) * rv_[q] * g0[c] + b0[c];
        ha[q * 128 + c] = f2bf(v);
    }
    __syncthreads();
    {   // out_1 = ln(x) @ Wl^T + bl ; x2 = tgt + out_1
        v8f acc = {};
#pragma unroll
        for (int kt = 0; kt < 4; ++kt) {
            v16bf a = load_a_frag(ha, 128, 0, kt * 32);
            v16bf bb = load_b_frag(WlBN, 128, wave * 16, kt * 32);
            acc = wmma_bf16(a, bb, acc);
        }
        int nn = lane & 15;
        int mb = (lane < 16) ? 0 : 8;
#pragma unroll
        for (int v = 0; v < 8; ++v) {
            int m = mb + v, col = wave * 16 + nn;
            xb_[m * 128 + col] = acc[v] + bl[col] + tgt[(size_t)(row0 + m) * 128 + col];
        }
    }
    __syncthreads();
    {   // second LN
        float s1 = 0.f, s2 = 0.f;
#pragma unroll
        for (int i = 0; i < 8; ++i) {
            float x = xb_[q * 128 + c0 + i];
            s1 += x; s2 += x * x;
        }
        red1[q][part] = s1; red2[q][part] = s2;
    }
    __syncthreads();
    if (t < 16) {
        float s1 = 0.f, s2 = 0.f;
#pragma unroll
        for (int i = 0; i < 16; ++i) { s1 += red1[t][i]; s2 += red2[t][i]; }
        float mu = s1 * (1.0f / 128.0f);
        float var = s2 * (1.0f / 128.0f) - mu * mu;
        mu_[t] = mu; rv_[t] = rsqrtf(var + 1e-5f);
    }
    __syncthreads();
#pragma unroll
    for (int i = 0; i < 8; ++i) {              // write [N, B, C]
        int c = c0 + i;
        float v = (xb_[q * 128 + c] - mu_[q]) * rv_[q] * g1[c] + b1[c];
        outg[((size_t)(n0 + q) * BATCH + b) * 128 + c] = v;
    }
}

// ---------------- host launch ----------------
extern "C" void kernel_launch(void* const* d_in, const int* in_sizes, int n_in,
                              void* d_out, int out_size, void* d_ws, size_t ws_size,
                              hipStream_t stream) {
    const float* src = (const float*)d_in[0];
    const float* tgt = (const float*)d_in[1];
    const float* Wq0 = (const float*)d_in[2];
    const float* Wk0 = (const float*)d_in[3];
    const float* Wv0 = (const float*)d_in[4];
    const float* Wq1 = (const float*)d_in[5];
    const float* Wk1 = (const float*)d_in[6];
    const float* Wv1 = (const float*)d_in[7];
    const float* Wp0 = (const float*)d_in[8];
    const float* bp0 = (const float*)d_in[9];
    const float* Wp1 = (const float*)d_in[10];
    const float* bp1 = (const float*)d_in[11];
    const float* Wl  = (const float*)d_in[12];
    const float* bl  = (const float*)d_in[13];
    const float* g0  = (const float*)d_in[14];
    const float* b0  = (const float*)d_in[15];
    const float* g1  = (const float*)d_in[16];
    const float* b1  = (const float*)d_in[17];

    char* w = (char*)d_ws;
    size_t o = 0;
    auto alloc = [&](size_t bytes) -> void* {
        o = (o + 255) & ~(size_t)255;
        void* p = w + o;
        o += bytes;
        return p;
    };
    int*            idx1   = (int*)alloc((size_t)ROWS * 16 * 4);
    int*            idx2   = (int*)alloc((size_t)ROWS * 16 * 4);
    float*          out1   = (float*)alloc((size_t)ROWS * 128 * 4);
    float*          out2   = (float*)alloc((size_t)ROWS * 128 * 4);
    float*          P1     = (float*)alloc((size_t)ROWS * 128 * 4);
    float*          P2     = (float*)alloc((size_t)ROWS * 128 * 4);
    unsigned short* out1bf = (unsigned short*)alloc((size_t)ROWS * 128 * 2);
    float*          r2     = (float*)alloc((size_t)ROWS * 4);
    unsigned short* M1t0   = (unsigned short*)alloc(128 * 128 * 2);
    unsigned short* M1t1   = (unsigned short*)alloc(128 * 128 * 2);
    unsigned short* Wv0b   = (unsigned short*)alloc(128 * 128 * 2);
    unsigned short* Wv1b   = (unsigned short*)alloc(128 * 128 * 2);
    unsigned short* Wlb    = (unsigned short*)alloc(128 * 128 * 2);
    unsigned short* Wp1b   = (unsigned short*)alloc(128 * 128 * 2);
    (void)in_sizes; (void)n_in; (void)out_size; (void)ws_size;

    prep_weights_kernel<<<384, 256, 0, stream>>>(Wq0, Wk0, Wq1, Wk1, Wv0, Wv1, Wl, Wp1,
                                                 M1t0, M1t1, Wv0b, Wv1b, Wlb, Wp1b);
    p1_kernel<<<ROWS * 128 / 256, 256, 0, stream>>>(src, Wp0, P1);
    knn_coord_kernel<<<ROWS / 256, 256, 0, stream>>>(src, idx1);
    attn_kernel<<<ROWS / 16, 256, 0, stream>>>(tgt, P1, idx1, M1t0, Wv0b, bp0, out1);
    cast_norm_kernel<<<ROWS / 256, 256, 0, stream>>>(out1, out1bf, r2);
    knn_feat_kernel<<<BATCH * (NPTS / 16), 256, 0, stream>>>(out1bf, r2, idx2);
    gemm_rows_kernel<<<ROWS / 64, 256, 0, stream>>>(out1, Wp1b, P2);
    attn_kernel<<<ROWS / 16, 256, 0, stream>>>(out1, P2, idx2, M1t1, Wv1b, bp1, out2);
    post_kernel<<<ROWS / 16, 256, 0, stream>>>(tgt, out2, g0, b0, Wlb, bl, g1, b1, (float*)d_out);
}
